// StructureModule_14035953123826
// MI455X (gfx1250) — matrix operand order repaired and checked by reading, not compiled, not run on hardware
//
#include <hip/hip_runtime.h>
#include <hip/hip_bf16.h>

typedef __attribute__((ext_vector_type(16))) _Float16 v16h;
typedef __attribute__((ext_vector_type(8)))  _Float16 v8h;
typedef __attribute__((ext_vector_type(8)))  float    v8f;

#define NRES 768
#define DIM  128
#define NH   12
#define PQK  4
#define PV   8

static inline int cdiv(int a, int b) { return (a + b - 1) / b; }

// ---------------------------------------------------------------------------
// A fragment from row-major f16: elements 0..15 = halfs [k, k+16)
// (p already has +16 for lanes 16..31).
// Two 16B vector loads, no conversion.
// ---------------------------------------------------------------------------
__device__ __forceinline__ v16h load_afrag16(const _Float16* __restrict__ p)
{
    v8h lo = *(const v8h*)(p);
    v8h hi = *(const v8h*)(p + 8);
    v16h f;
#pragma unroll
    for (int i = 0; i < 8; ++i) { f[i] = lo[i]; f[8 + i] = hi[i]; }
    return f;
}

// ---------------------------------------------------------------------------
// Fast WMMA GEMM, all-f16 operands, no bounds checks, no in-loop conversion.
// One wave computes a 64(M) x 16(N) tile: 4 m-subtiles share one B fragment.
// Per K-step: load B + 4 A fragments into distinct registers, then issue the
// 4 WMMAs (lets waits cover a whole step and overlap with the next one).
// BMODE 0: B pre-packed in fragment layout (16 halfs/lane/K-step, see pack_b).
// BMODE 1: B row-major f16 addressed transposed: (k,n) = B[n*ldb + k].
// C (f32) and C16 (f16 row-major) outputs are each optional (nullptr skips).
// Requirements: M = gridDim.y*64, N = gridDim.x*16, K multiple of 32.
// ---------------------------------------------------------------------------
template <int BMODE, bool RELU>
__global__ __launch_bounds__(32) void wmma_gemm_fast(
    const _Float16* __restrict__ A, long lda, long a_hs,
    const _Float16* __restrict__ B, long ldb, long b_hs,
    const float* __restrict__ bias,
    float* __restrict__ C, long ldc, long c_hs,
    _Float16* __restrict__ C16, long ldc16, long c16_hs,
    int K, float alpha)
{
    const int lane = threadIdx.x;
    const int h    = blockIdx.z;
    const int m0   = blockIdx.y * 64;
    const int n0   = blockIdx.x * 16;
    const int half = lane >> 4;
    const int lr   = lane & 15;
    const int KS   = K >> 5;

    A += (long)h * a_hs;
    const _Float16* a0p = A + (long)(m0 +  0 + lr) * lda + half * 16;
    const _Float16* a1p = A + (long)(m0 + 16 + lr) * lda + half * 16;
    const _Float16* a2p = A + (long)(m0 + 32 + lr) * lda + half * 16;
    const _Float16* a3p = A + (long)(m0 + 48 + lr) * lda + half * 16;

    const _Float16* bp;
    if (BMODE == 0)
        bp = B + ((((long)h * gridDim.x + blockIdx.x) * KS) * 32 + lane) * 16;
    else
        bp = B + (long)h * b_hs + (long)(n0 + lr) * ldb + half * 16;

    v8f acc0 = {}, acc1 = {}, acc2 = {}, acc3 = {};

    for (int k0 = 0; k0 < K; k0 += 32) {
        v16h bf = *(const v16h*)bp;
        bp += (BMODE == 0) ? 512 : 32;
        v16h af0 = load_afrag16(a0p); a0p += 32;
        v16h af1 = load_afrag16(a1p); a1p += 32;
        v16h af2 = load_afrag16(a2p); a2p += 32;
        v16h af3 = load_afrag16(a3p); a3p += 32;
        acc0 = __builtin_amdgcn_wmma_f32_16x16x32_f16(false, af0, false, bf, (short)0, acc0, false, false);
        acc1 = __builtin_amdgcn_wmma_f32_16x16x32_f16(false, af1, false, bf, (short)0, acc1, false, false);
        acc2 = __builtin_amdgcn_wmma_f32_16x16x32_f16(false, af2, false, bf, (short)0, acc2, false, false);
        acc3 = __builtin_amdgcn_wmma_f32_16x16x32_f16(false, af3, false, bf, (short)0, acc3, false, false);
    }

    const float bv = bias ? bias[n0 + lr] : 0.0f;
    if (C)   C   += (long)h * c_hs   + n0 + lr;
    if (C16) C16 += (long)h * c16_hs + n0 + lr;
#pragma unroll
    for (int t = 0; t < 4; ++t) {
        const v8f* at = (t == 0) ? &acc0 : (t == 1) ? &acc1 : (t == 2) ? &acc2 : &acc3;
#pragma unroll
        for (int r = 0; r < 8; ++r) {
            long row = m0 + t * 16 + r + 8 * half;
            float v = (*at)[r] * alpha + bv;
            if (RELU) v = fmaxf(v, 0.0f);
            if (C)   C[row * ldc] = v;
            if (C16) C16[row * ldc16] = (_Float16)v;
        }
    }
}

// ---------------------------------------------------------------------------
// Pack B[K][N] (row stride ldb, head stride b_hs, source f32 or f16) into the
// WMMA B-fragment layout consumed by BMODE==0:
//   frag idx = ((h*NT + tn)*KS + ks)*32 + lane ; 16 halfs each,
//   element i -> B[(ks*32 + i + 16*(lane>=16))*ldb + tn*16 + (lane&15)]
// ---------------------------------------------------------------------------
template <typename T>
__global__ void pack_b_kernel(const T* __restrict__ B, long ldb, long b_hs,
                              _Float16* __restrict__ Bp, int K, int NT, int total)
{
    int idx = blockIdx.x * blockDim.x + threadIdx.x;
    if (idx >= total) return;
    int KS   = K >> 5;
    int lane = idx & 31;
    int rest = idx >> 5;          // (h*NT+tn)*KS + ks
    int ks   = rest % KS;
    int tnh  = rest / KS;
    int tn   = tnh % NT;
    int h    = tnh / NT;

    const T* src = B + (long)h * b_hs + (long)(tn * 16 + (lane & 15));
    long kbase = (long)ks * 32 + ((lane >= 16) ? 16 : 0);
    v16h f;
#pragma unroll
    for (int i = 0; i < 16; ++i) f[i] = (_Float16)src[(kbase + i) * ldb];
    *(v16h*)(Bp + (long)idx * 16) = f;
}

// ---------------------------------------------------------------------------
// Backbone linear (768 x 6 x 128) in plain f32 — too ragged for WMMA tiles.
// ---------------------------------------------------------------------------
__global__ void bb_linear_kernel(const float* __restrict__ x,
                                 const float* __restrict__ w,
                                 const float* __restrict__ b,
                                 float* __restrict__ u, int N)
{
    int idx = blockIdx.x * blockDim.x + threadIdx.x;
    if (idx >= N * 6) return;
    int n = idx / 6, j = idx - n * 6;
    float s = b[j];
    const float* xr = x + (long)n * DIM;
#pragma unroll 8
    for (int k = 0; k < DIM; ++k) s += xr[k] * w[k * 6 + j];
    u[(long)n * 8 + j] = s;
}

// ---------------------------------------------------------------------------
// Rigid frames from 3 backbone atoms (Gram-Schmidt). coords: [N][3][3]
// ---------------------------------------------------------------------------
__global__ void frames_kernel(const float* __restrict__ coords,
                              float* __restrict__ R, float* __restrict__ t, int N)
{
    int n = blockIdx.x * blockDim.x + threadIdx.x;
    if (n >= N) return;
    const float* c = coords + (long)n * 9;
    float pnx[3] = { c[0], c[1], c[2] };
    float org[3] = { c[3], c[4], c[5] };
    float pxy[3] = { c[6], c[7], c[8] };

    float e0[3], e1[3], e2[3];
    for (int i = 0; i < 3; ++i) e0[i] = org[i] - pnx[i];
    float n0 = sqrtf(e0[0]*e0[0] + e0[1]*e0[1] + e0[2]*e0[2]) + 1e-8f;
    for (int i = 0; i < 3; ++i) e0[i] /= n0;
    for (int i = 0; i < 3; ++i) e1[i] = pxy[i] - org[i];
    float dot = e0[0]*e1[0] + e0[1]*e1[1] + e0[2]*e1[2];
    for (int i = 0; i < 3; ++i) e1[i] -= e0[i] * dot;
    float n1 = sqrtf(e1[0]*e1[0] + e1[1]*e1[1] + e1[2]*e1[2]) + 1e-8f;
    for (int i = 0; i < 3; ++i) e1[i] /= n1;
    e2[0] = e0[1]*e1[2] - e0[2]*e1[1];
    e2[1] = e0[2]*e1[0] - e0[0]*e1[2];
    e2[2] = e0[0]*e1[1] - e0[1]*e1[0];

    float* Rn = R + (long)n * 9;
    for (int i = 0; i < 3; ++i) {
        Rn[i*3 + 0] = e0[i];
        Rn[i*3 + 1] = e1[i];
        Rn[i*3 + 2] = e2[i];
    }
    float* tn = t + (long)n * 3;
    tn[0] = org[0]; tn[1] = org[1]; tn[2] = org[2];
}

// ---------------------------------------------------------------------------
// LayerNorm over D=128 (+optional residual), writing f32 and f16 copies.
// ---------------------------------------------------------------------------
__global__ __launch_bounds__(128) void ln_kernel(
    const float* __restrict__ X, const float* __restrict__ Res,
    const float* __restrict__ g, const float* __restrict__ b,
    float* __restrict__ Out32, _Float16* __restrict__ Out16)
{
    __shared__ float red[128];
    int row = blockIdx.x, c = threadIdx.x;
    float v = X[(long)row * DIM + c];
    if (Res) v += Res[(long)row * DIM + c];

    red[c] = v; __syncthreads();
    for (int s = 64; s > 0; s >>= 1) { if (c < s) red[c] += red[c + s]; __syncthreads(); }
    float mu = red[0] / (float)DIM; __syncthreads();

    float d = v - mu;
    red[c] = d * d; __syncthreads();
    for (int s = 64; s > 0; s >>= 1) { if (c < s) red[c] += red[c + s]; __syncthreads(); }
    float var = red[0] / (float)DIM;

    float y = d * rsqrtf(var + 1e-5f) * g[c] + b[c];
    Out32[(long)row * DIM + c] = y;
    Out16[(long)row * DIM + c] = (_Float16)y;
}

// ---------------------------------------------------------------------------
// Rotate projected points into global frame: pts[n][j][.] = R[n]*v + t[n]
// ---------------------------------------------------------------------------
__global__ void rot_points_kernel(const float* __restrict__ lin,
                                  const float* __restrict__ R,
                                  const float* __restrict__ t,
                                  float* __restrict__ pts, int N, int J)
{
    int idx = blockIdx.x * blockDim.x + threadIdx.x;
    if (idx >= N * J) return;
    int n = idx / J, j = idx - n * J;
    const float* ln = lin + (long)n * 3 * J;
    float vx = ln[0 * J + j], vy = ln[1 * J + j], vz = ln[2 * J + j];
    const float* Rn = R + (long)n * 9;
    const float* tn = t + (long)n * 3;
    float* p = pts + ((long)n * J + j) * 3;
    p[0] = Rn[0]*vx + Rn[1]*vy + Rn[2]*vz + tn[0];
    p[1] = Rn[3]*vx + Rn[4]*vy + Rn[5]*vz + tn[1];
    p[2] = Rn[6]*vx + Rn[7]*vy + Rn[8]*vz + tn[2];
}

// ---------------------------------------------------------------------------
// Fused point-distance bias + mask + softmax over one attention row (h,i).
// Reads f32 logits, writes normalized weights as f16 (GEMM operand).
// ---------------------------------------------------------------------------
__global__ __launch_bounds__(256) void softmax_pt_kernel(
    const float* __restrict__ att, _Float16* __restrict__ atth,
    const float* __restrict__ qpts,   // [N][H*PQK][3]
    const float* __restrict__ kvpts,  // [N][H*(PQK+PV)][3]
    const float* __restrict__ mask,   // [N]
    const float* __restrict__ head_weights, int N)
{
    __shared__ float qp_s[PQK * 3];
    __shared__ float red[256];
    int i = blockIdx.x, h = blockIdx.y, tid = threadIdx.x;
    const float* row = att + ((long)h * N + i) * (long)N;
    _Float16* row16 = atth + ((long)h * N + i) * (long)N;

    if (tid < PQK * 3) qp_s[tid] = qpts[(long)i * (NH*PQK*3) + h * (PQK*3) + tid];
    __syncthreads();

    float hw = log1pf(expf(head_weights[h])) * 0.13608276348795434f; // softplus * sqrt(1/54)
    float maski = mask[i];

    float vals[3];
    float mx = -3.402823466e+38f;
#pragma unroll
    for (int e = 0; e < 3; ++e) {
        int j = tid + e * 256;
        const float* kp = kvpts + (long)j * (NH*(PQK+PV)*3) + h * ((PQK+PV)*3);
        float s = 0.0f;
#pragma unroll
        for (int pc = 0; pc < PQK * 3; ++pc) {
            float d = qp_s[pc] - kp[pc];
            s += d * d;
        }
        float v = row[j] - 0.5f * hw * s + 1e5f * (maski * mask[j] - 1.0f);
        vals[e] = v;
        mx = fmaxf(mx, v);
    }
    red[tid] = mx; __syncthreads();
    for (int s = 128; s > 0; s >>= 1) { if (tid < s) red[tid] = fmaxf(red[tid], red[tid + s]); __syncthreads(); }
    mx = red[0]; __syncthreads();

    float sum = 0.0f;
#pragma unroll
    for (int e = 0; e < 3; ++e) { vals[e] = expf(vals[e] - mx); sum += vals[e]; }
    red[tid] = sum; __syncthreads();
    for (int s = 128; s > 0; s >>= 1) { if (tid < s) red[tid] += red[tid + s]; __syncthreads(); }
    float inv = 1.0f / red[0];
#pragma unroll
    for (int e = 0; e < 3; ++e) row16[tid + e * 256] = (_Float16)(vals[e] * inv);
}

// ---------------------------------------------------------------------------
// o_pt finalize: local = R^T * global - t ; write xyz + norm into feat (f16).
// optr: [N][H*32] f32 (valid cols p*3+c in [0,24) per head).
// ---------------------------------------------------------------------------
__global__ void finalize_opt_kernel(const float* __restrict__ optr,
                                    const float* __restrict__ R,
                                    const float* __restrict__ t,
                                    _Float16* __restrict__ feat, int N)
{
    int idx = blockIdx.x * blockDim.x + threadIdx.x;
    if (idx >= N * NH * PV) return;
    int n = idx / (NH * PV);
    int hp = idx - n * (NH * PV);
    int h = hp / PV, p = hp - h * PV;

    const float* g  = optr + (long)n * (NH * 32) + h * 32 + p * 3;
    const float* Rn = R + (long)n * 9;
    const float* tn = t + (long)n * 3;
    float gx = g[0], gy = g[1], gz = g[2];
    float lx = Rn[0]*gx + Rn[3]*gy + Rn[6]*gz - tn[0];
    float ly = Rn[1]*gx + Rn[4]*gy + Rn[7]*gz - tn[1];
    float lz = Rn[2]*gx + Rn[5]*gy + Rn[8]*gz - tn[2];
    float nr = sqrtf(lx*lx + ly*ly + lz*lz + 1e-8f);

    _Float16* f = feat + (long)n * 1920 + NH * DIM;   // col 1536
    f[0 * 96 + hp] = (_Float16)lx;
    f[1 * 96 + hp] = (_Float16)ly;
    f[2 * 96 + hp] = (_Float16)lz;
    f[3 * 96 + hp] = (_Float16)nr;
}

// ---------------------------------------------------------------------------
// Backbone update: quaternion -> R_u; R = R @ R_u; t = t + R_new @ t_u
// ---------------------------------------------------------------------------
__global__ void backbone_kernel(const float* __restrict__ ubuf,
                                float* __restrict__ R, float* __restrict__ t, int N)
{
    int n = blockIdx.x * blockDim.x + threadIdx.x;
    if (n >= N) return;
    const float* u = ubuf + (long)n * 8;
    float b = u[0], c = u[1], d = u[2];
    float norm = sqrtf(1.0f + b*b + c*c + d*d);
    float a = 1.0f / norm; b /= norm; c /= norm; d /= norm;
    float a2=a*a, b2=b*b, c2=c*c, d2=d*d;
    float ab=a*b, ac=a*c, ad=a*d, bc=b*c, bd=b*d, cd=c*d;
    float Ru[9] = { a2+b2-c2-d2, 2*(bc-ad),   2*(bd+ac),
                    2*(bc+ad),   a2-b2+c2-d2, 2*(cd-ab),
                    2*(bd-ac),   2*(cd+ab),   a2-b2-c2+d2 };
    float* Rn = R + (long)n * 9;
    float* tn = t + (long)n * 3;
    float Ro[9];
#pragma unroll
    for (int i = 0; i < 9; ++i) Ro[i] = Rn[i];
    float Rnew[9];
#pragma unroll
    for (int i = 0; i < 3; ++i)
#pragma unroll
        for (int j = 0; j < 3; ++j)
            Rnew[i*3+j] = Ro[i*3+0]*Ru[0*3+j] + Ro[i*3+1]*Ru[1*3+j] + Ro[i*3+2]*Ru[2*3+j];
    float tu[3] = { u[3], u[4], u[5] };
#pragma unroll
    for (int i = 0; i < 9; ++i) Rn[i] = Rnew[i];
#pragma unroll
    for (int i = 0; i < 3; ++i)
        tn[i] += Rnew[i*3+0]*tu[0] + Rnew[i*3+1]*tu[1] + Rnew[i*3+2]*tu[2];
}

// ---------------------------------------------------------------------------
// Host orchestration
// ---------------------------------------------------------------------------
extern "C" void kernel_launch(void* const* d_in, const int* in_sizes, int n_in,
                              void* d_out, int out_size, void* d_ws, size_t ws_size,
                              hipStream_t stream)
{
    (void)in_sizes; (void)n_in; (void)out_size; (void)ws_size;
    const int N = NRES;

    const float* embeddings   = (const float*)d_in[0];
    const float* coords       = (const float*)d_in[1];
    const float* mask         = (const float*)d_in[2];
    const float* input_ln_g   = (const float*)d_in[3];
    const float* input_ln_b   = (const float*)d_in[4];
    const float* input_fc_w   = (const float*)d_in[5];
    const float* input_fc_b   = (const float*)d_in[6];
    const float* q_w          = (const float*)d_in[7];
    const float* q_b          = (const float*)d_in[8];
    const float* kv_w         = (const float*)d_in[9];
    const float* kv_b         = (const float*)d_in[10];
    const float* qp_w         = (const float*)d_in[11];
    const float* qp_b         = (const float*)d_in[12];
    const float* kvp_w        = (const float*)d_in[13];
    const float* kvp_b        = (const float*)d_in[14];
    const float* head_weights = (const float*)d_in[15];
    const float* out_w        = (const float*)d_in[16];
    const float* out_b        = (const float*)d_in[17];
    const float* ipa_ln_g     = (const float*)d_in[18];
    const float* ipa_ln_b     = (const float*)d_in[19];
    const float* t1_w         = (const float*)d_in[20];
    const float* t1_b         = (const float*)d_in[21];
    const float* t2_w         = (const float*)d_in[22];
    const float* t2_b         = (const float*)d_in[23];
    const float* trans_ln_g   = (const float*)d_in[24];
    const float* trans_ln_b   = (const float*)d_in[25];
    const float* bb_w         = (const float*)d_in[26];
    const float* bb_b         = (const float*)d_in[27];

    // Workspace carve-up (all offsets multiples of 8 floats = 32B)
    float* ws = (float*)d_ws;
    float* Rf     = ws;  ws += (long)N * 9;
    float* tf     = ws;  ws += (long)N * 3;
    float* xn     = ws;  ws += (long)N * DIM;
    _Float16* xnh = (_Float16*)ws;  ws += (long)N * DIM / 2;
    float* x      = ws;  ws += (long)N * DIM;
    _Float16* xh  = (_Float16*)ws;  ws += (long)N * DIM / 2;
    _Float16* qh  = (_Float16*)ws;  ws += (long)N * (NH*DIM) / 2;
    _Float16* kvh = (_Float16*)ws;  ws += (long)N * (NH*2*DIM) / 2;
    float* qplin  = ws;  ws += (long)N * 144;
    float* qpts   = ws;  ws += (long)N * 144;
    float* kvlin  = ws;  ws += (long)N * 432;
    float* kvpts  = ws;  ws += (long)N * 432 + 64;   // pad for o_pt N=24->32 over-read
    float* att    = ws;  ws += (long)NH * N * N;
    _Float16* atth= (_Float16*)ws;  ws += (long)NH * N * N / 2;
    _Float16* feat= (_Float16*)ws;  ws += (long)N * 1920 / 2;
    float* optr   = ws;  ws += (long)N * (NH * 32);
    float* iout   = ws;  ws += (long)N * DIM;
    float* hb2    = ws;  ws += (long)N * DIM;
    _Float16* h1h = (_Float16*)ws;  ws += (long)N * DIM / 2;
    float* ubuf   = ws;  ws += (long)N * 8;
    // packed weight fragments (halfs)
    _Float16* fcP  = (_Float16*)ws; ws += (long)DIM * DIM / 2;
    _Float16* qP   = (_Float16*)ws; ws += (long)DIM * (NH*DIM) / 2;
    _Float16* kvP  = (_Float16*)ws; ws += (long)DIM * (NH*2*DIM) / 2;
    _Float16* qpP  = (_Float16*)ws; ws += (long)DIM * 144 / 2;
    _Float16* kvpP = (_Float16*)ws; ws += (long)DIM * 432 / 2;
    _Float16* outP = (_Float16*)ws; ws += (long)1920 * DIM / 2;
    _Float16* t1P  = (_Float16*)ws; ws += (long)DIM * DIM / 2;
    _Float16* t2P  = (_Float16*)ws; ws += (long)DIM * DIM / 2;
    _Float16* vP   = (_Float16*)ws; ws += (long)NH * DIM * N / 2;   // 12 heads x (768x128)
    _Float16* vpP  = (_Float16*)ws; ws += (long)NH * 32 * N / 2;    // 12 heads x (768x32)

    const float logit_scale = 0.05103103630798287f; // sqrt(1/(3*128))
    const int MT = N / 64;

    // --- one-time: frames, weight packing, input LN + fc ---
    frames_kernel<<<cdiv(N, 256), 256, 0, stream>>>(coords, Rf, tf, N);

    {   // pack all weight matrices (f32 [K][N] -> fragment f16)
        int tot;
        tot = 1*(DIM/16)*(DIM/32)*32;
        pack_b_kernel<float><<<cdiv(tot,256),256,0,stream>>>(input_fc_w, DIM, 0, fcP, DIM, DIM/16, tot);
        pack_b_kernel<float><<<cdiv(tot,256),256,0,stream>>>(t1_w, DIM, 0, t1P, DIM, DIM/16, tot);
        pack_b_kernel<float><<<cdiv(tot,256),256,0,stream>>>(t2_w, DIM, 0, t2P, DIM, DIM/16, tot);
        tot = 1*((NH*DIM)/16)*(DIM/32)*32;
        pack_b_kernel<float><<<cdiv(tot,256),256,0,stream>>>(q_w, NH*DIM, 0, qP, DIM, (NH*DIM)/16, tot);
        tot = 1*((NH*2*DIM)/16)*(DIM/32)*32;
        pack_b_kernel<float><<<cdiv(tot,256),256,0,stream>>>(kv_w, NH*2*DIM, 0, kvP, DIM, (NH*2*DIM)/16, tot);
        tot = 1*(144/16)*(DIM/32)*32;
        pack_b_kernel<float><<<cdiv(tot,256),256,0,stream>>>(qp_w, 144, 0, qpP, DIM, 144/16, tot);
        tot = 1*(432/16)*(DIM/32)*32;
        pack_b_kernel<float><<<cdiv(tot,256),256,0,stream>>>(kvp_w, 432, 0, kvpP, DIM, 432/16, tot);
        tot = 1*(DIM/16)*(1920/32)*32;
        pack_b_kernel<float><<<cdiv(tot,256),256,0,stream>>>(out_w, DIM, 0, outP, 1920, DIM/16, tot);
    }

    ln_kernel<<<N, 128, 0, stream>>>(embeddings, nullptr, input_ln_g, input_ln_b, xn, xnh);
    wmma_gemm_fast<0, false><<<dim3(DIM/16, MT, 1), 32, 0, stream>>>(
        xnh, DIM, 0, fcP, 0, 0, input_fc_b,
        x, DIM, 0, xh, DIM, 0, DIM, 1.0f);

    for (int layer = 0; layer < 2; ++layer) {
        // projections (f16 outputs feed attention; qp/kvp stay f32 for geometry)
        wmma_gemm_fast<0, false><<<dim3((NH*DIM)/16, MT, 1), 32, 0, stream>>>(
            xh, DIM, 0, qP, 0, 0, q_b,
            nullptr, 0, 0, qh, NH*DIM, 0, DIM, 1.0f);
        wmma_gemm_fast<0, false><<<dim3((NH*2*DIM)/16, MT, 1), 32, 0, stream>>>(
            xh, DIM, 0, kvP, 0, 0, kv_b,
            nullptr, 0, 0, kvh, NH*2*DIM, 0, DIM, 1.0f);
        wmma_gemm_fast<0, false><<<dim3(144/16, MT, 1), 32, 0, stream>>>(
            xh, DIM, 0, qpP, 0, 0, qp_b,
            qplin, 144, 0, nullptr, 0, 0, DIM, 1.0f);
        wmma_gemm_fast<0, false><<<dim3(432/16, MT, 1), 32, 0, stream>>>(
            xh, DIM, 0, kvpP, 0, 0, kvp_b,
            kvlin, 432, 0, nullptr, 0, 0, DIM, 1.0f);

        rot_points_kernel<<<cdiv(N*48, 256), 256, 0, stream>>>(qplin, Rf, tf, qpts, N, 48);
        rot_points_kernel<<<cdiv(N*144, 256), 256, 0, stream>>>(kvlin, Rf, tf, kvpts, N, 144);

        // logits: att[h] = scale * q_h @ k_h^T  (BT over row-major f16 kv)
        wmma_gemm_fast<1, false><<<dim3(N/16, MT, NH), 32, 0, stream>>>(
            qh, NH*DIM, DIM,
            kvh, NH*2*DIM, 2*DIM,
            nullptr,
            att, N, (long)N*N, nullptr, 0, 0, DIM, logit_scale);

        softmax_pt_kernel<<<dim3(N, NH), 256, 0, stream>>>(
            att, atth, qpts, kvpts, mask, head_weights, N);

        // pack v (from f16 kv) and vp (from f32 global points, padded to 32 cols)
        {
            int tot = NH*(DIM/16)*(N/32)*32;
            pack_b_kernel<_Float16><<<cdiv(tot,256),256,0,stream>>>(
                kvh + DIM, NH*2*DIM, 2*DIM, vP, N, DIM/16, tot);
            tot = NH*2*(N/32)*32;
            pack_b_kernel<float><<<cdiv(tot,256),256,0,stream>>>(
                kvpts + PQK*3, 432, (PQK+PV)*3, vpP, N, 2, tot);
        }

        // o = a @ v -> feat[:, h*128:(h+1)*128] (f16)
        wmma_gemm_fast<0, false><<<dim3(DIM/16, MT, NH), 32, 0, stream>>>(
            atth, N, (long)N*N, vP, 0, 0, nullptr,
            nullptr, 0, 0, feat, 1920, DIM, N, 1.0f);

        // o_pt = a @ vp (N padded 24->32; cols 24..31 garbage, never read)
        wmma_gemm_fast<0, false><<<dim3(2, MT, NH), 32, 0, stream>>>(
            atth, N, (long)N*N, vpP, 0, 0, nullptr,
            optr, NH*32, 32, nullptr, 0, 0, N, 1.0f);

        finalize_opt_kernel<<<cdiv(N*NH*PV, 256), 256, 0, stream>>>(optr, Rf, tf, feat, N);

        // out projection + residual LN
        wmma_gemm_fast<0, false><<<dim3(DIM/16, MT, 1), 32, 0, stream>>>(
            feat, 1920, 0, outP, 0, 0, out_b,
            iout, DIM, 0, nullptr, 0, 0, 1920, 1.0f);
        ln_kernel<<<N, 128, 0, stream>>>(x, iout, ipa_ln_g, ipa_ln_b, x, xh);

        // transition MLP + residual LN
        wmma_gemm_fast<0, true><<<dim3(DIM/16, MT, 1), 32, 0, stream>>>(
            xh, DIM, 0, t1P, 0, 0, t1_b,
            nullptr, 0, 0, h1h, DIM, 0, DIM, 1.0f);
        wmma_gemm_fast<0, false><<<dim3(DIM/16, MT, 1), 32, 0, stream>>>(
            h1h, DIM, 0, t2P, 0, 0, t2_b,
            hb2, DIM, 0, nullptr, 0, 0, DIM, 1.0f);

        float* ln_out = (layer == 1) ? (float*)d_out : x;
        ln_kernel<<<N, 128, 0, stream>>>(x, hb2, trans_ln_g, trans_ln_b, ln_out, xh);

        if (layer == 0) {
            bb_linear_kernel<<<cdiv(N*6, 256), 256, 0, stream>>>(x, bb_w, bb_b, ubuf, N);
            backbone_kernel<<<cdiv(N, 256), 256, 0, stream>>>(ubuf, Rf, tf, N);
        }
    }
}
